// RotationGeoDynamicLayer_37048387895888
// MI455X (gfx1250) — matrix operand order repaired and checked by reading, not compiled, not run on hardware
//
#include <hip/hip_runtime.h>
#include <math.h>

typedef __attribute__((ext_vector_type(16))) __bf16 v16bf;
typedef __attribute__((ext_vector_type(8)))  float  v8f;
typedef unsigned short u16;
typedef unsigned int   u32;

#define D_   768
#define B_   32
#define N_   512
#define HID_ 128
#define K_   8
#define R_   8
#define ZD   2304      // 3*D
#define RNK  128       // 2*K*R : rank of low-rank skew factorization
#define MTOT 16384     // B*N

#define RESIDUAL_SCALE_ 1.0f
#define ADAPTER_SCALE_  1.0f

// ---------- workspace layout (bytes, all 256-aligned) ----------
constexpr size_t OFF_XBF  = 0;                              // x in bf16 (16384x768)
constexpr size_t SZ_XBF   = (size_t)MTOT * D_ * 2;
constexpr size_t OFF_Z    = OFF_XBF + SZ_XBF;               // z (32x2304) f32
constexpr size_t SZ_Z     = (size_t)B_ * ZD * 4;
constexpr size_t OFF_H    = OFF_Z + SZ_Z;                   // h (32x128) f32
constexpr size_t SZ_H     = (size_t)B_ * HID_ * 4;
constexpr size_t OFF_NRM  = OFF_H + SZ_H;                   // norms (8) f32
constexpr size_t SZ_NRM   = 256;
constexpr size_t OFF_GATE = OFF_NRM + SZ_NRM;               // gate (32) f32
constexpr size_t SZ_GATE  = 256;
constexpr size_t OFF_S    = OFF_GATE + SZ_GATE;             // s (32x8) f32
constexpr size_t SZ_S     = 1024;
constexpr size_t OFF_AF   = OFF_S + SZ_S;                   // A_base f32 (768x128)
constexpr size_t SZ_AF    = (size_t)D_ * RNK * 4;
constexpr size_t OFF_BF   = OFF_AF + SZ_AF;                 // B f32 (768x128)
constexpr size_t SZ_BF    = (size_t)D_ * RNK * 4;
constexpr size_t OFF_ABT  = OFF_BF + SZ_BF;                 // A_base^T bf16 (128x768)
constexpr size_t SZ_ABT   = (size_t)RNK * D_ * 2;
constexpr size_t OFF_BBT  = OFF_ABT + SZ_ABT;               // B^T bf16 (128x768)
constexpr size_t SZ_BBT   = (size_t)RNK * D_ * 2;
constexpr size_t OFF_WBF  = OFF_BBT + SZ_BBT;               // W bf16 (768x768 row-major)
constexpr size_t SZ_WBF   = (size_t)D_ * D_ * 2;
constexpr size_t OFF_P    = OFF_WBF + SZ_WBF;               // P = B^T A_base f32 (128x128)
constexpr size_t SZ_P     = (size_t)RNK * RNK * 4;
constexpr size_t OFF_MIT  = OFF_P + SZ_P;                   // Minv''^T bf16 (32x128x128)
constexpr size_t SZ_MIT   = (size_t)B_ * RNK * RNK * 2;
constexpr size_t OFF_ZBF  = OFF_MIT + SZ_MIT;               // Z = x@A bf16 (16384x128)
constexpr size_t SZ_ZBF   = (size_t)MTOT * RNK * 2;
constexpr size_t OFF_YBF  = OFF_ZBF + SZ_ZBF;               // Y bf16 (16384x128)
constexpr size_t SZ_YBF   = (size_t)MTOT * RNK * 2;
constexpr size_t OFF_CBF  = OFF_YBF + SZ_YBF;               // C = W@B bf16 (768x128)
constexpr size_t SZ_CBF   = (size_t)D_ * RNK * 2;

// ---------- helpers ----------
__device__ inline u16 f2bf(float f) {
    union { float f; u32 u; } v; v.f = f;
    u32 r = v.u + 0x7FFFu + ((v.u >> 16) & 1u);   // round-to-nearest-even
    return (u16)(r >> 16);
}

union V16 { v16bf v; uint4 q[2]; };

// A-operand fragment (16x32 bf16, row-major source, ld in elements).
// ISA layout: lanes 0-15 hold M=lane, K = k0+[0..7] and k0+16+[0..7];
//             lanes 16-31 hold M=lane-16, K shifted by +8.
__device__ inline v16bf load_a16(const u16* __restrict__ base, int ld, int m0, int k0) {
    int lane = threadIdx.x & 31;
    int hi   = lane >> 4;
    int m    = m0 + (lane & 15);
    const u16* p = base + (size_t)m * ld + k0 + hi * 8;
    V16 r;
    r.q[0] = *(const uint4*)(p);        // K = k0+off .. +7
    r.q[1] = *(const uint4*)(p + 16);   // K = k0+16+off .. +7
    return r.v;
}

// B-operand fragment (32x16 bf16) read from a TRANSPOSED source baseT[n][k]
// (so each lane does contiguous 16B loads): lanes 0-15 K=k0+[0..7] and k0+16+[0..7] col=lane,
// lanes 16-31 K shifted by +8, col=lane-16.
__device__ inline v16bf load_bT16(const u16* __restrict__ baseT, int ld, int k0, int n0) {
    int lane = threadIdx.x & 31;
    int n = n0 + (lane & 15);
    int k = k0 + ((lane >> 4) << 3);
    const u16* p = baseT + (size_t)n * ld + k;
    V16 r;
    r.q[0] = *(const uint4*)(p);
    r.q[1] = *(const uint4*)(p + 16);
    return r.v;
}

// ---------- K1: stats (cls/mean/var -> z) + x -> bf16 ----------
__global__ __launch_bounds__(256) void k_stats(const float* __restrict__ x,
                                               float* __restrict__ z,
                                               u16* __restrict__ xbf) {
    int idx = blockIdx.x * 256 + threadIdx.x;   // B_*D_ threads
    int b = idx / D_, d = idx % D_;
    const float* xp = x + (size_t)b * N_ * D_ + d;
    u16* xo = xbf + (size_t)b * N_ * D_ + d;
    float s = 0.f, sq = 0.f, cls = 0.f;
    for (int n = 0; n < N_; ++n) {
        float v = xp[(size_t)n * D_];
        if (n == 0) cls = v;
        s += v; sq = fmaf(v, v, sq);
        xo[(size_t)n * D_] = f2bf(v);
    }
    float mean = s * (1.0f / N_);
    float var  = sq * (1.0f / N_) - mean * mean;
    z[b * ZD + d]          = cls;
    z[b * ZD + D_ + d]     = mean;
    z[b * ZD + 2 * D_ + d] = var;
}

// ---------- K2: build A_base=[L|-R], B=[R|L] (f32 + transposed bf16) ----------
__global__ __launch_bounds__(256) void k_build(const float* __restrict__ lb,
                                               const float* __restrict__ rb,
                                               float* __restrict__ Af, float* __restrict__ Bf,
                                               u16* __restrict__ AbT, u16* __restrict__ BbT) {
    int idx = blockIdx.x * 256 + threadIdx.x;   // D_*RNK threads
    int d = idx / RNK, j = idx % RNK;
    int jj = j & 63, k = jj >> 3, r = jj & 7;
    float lv = lb[((size_t)k * D_ + d) * R_ + r];
    float rv = rb[((size_t)k * D_ + d) * R_ + r];
    float a = (j < 64) ? lv : -rv;
    float bb = (j < 64) ? rv : lv;
    Af[(size_t)d * RNK + j] = a;
    Bf[(size_t)d * RNK + j] = bb;
    AbT[(size_t)j * D_ + d] = f2bf(a);
    BbT[(size_t)j * D_ + d] = f2bf(bb);
}

// ---------- K3: W -> bf16 ----------
__global__ __launch_bounds__(256) void k_wbf(const float* __restrict__ w, u16* __restrict__ wb) {
    int i = blockIdx.x * 256 + threadIdx.x;
    wb[i] = f2bf(w[i]);
}

// ---------- K4: basis Frobenius norms via 8x8 Gram matrices ----------
// ||L R^T - R L^T||_F^2 = 2 tr(Gll Grr) - 2 tr(Glr^2)
__global__ __launch_bounds__(256) void k_norm(const float* __restrict__ lb,
                                              const float* __restrict__ rb,
                                              float* __restrict__ norms) {
    __shared__ float g[192];
    __shared__ float tr[64];
    int k = blockIdx.x, t = threadIdx.x;
    if (t < 192) {
        int which = t / 64, e = t % 64, a = e >> 3, bcol = e & 7;
        const float* U = lb + (size_t)k * D_ * R_;
        const float* V = rb + (size_t)k * D_ * R_;
        const float* P0 = (which == 1) ? V : U;
        const float* P1 = (which == 0) ? U : V;
        float acc = 0.f;
        for (int d = 0; d < D_; ++d) acc = fmaf(P0[d * R_ + a], P1[d * R_ + bcol], acc);
        g[t] = acc;   // [0..63]=Gll, [64..127]=Grr, [128..191]=Glr
    }
    __syncthreads();
    if (t < 64) {
        int a = t >> 3, b = t & 7;
        tr[t] = g[a * 8 + b] * g[64 + b * 8 + a] - g[128 + a * 8 + b] * g[128 + b * 8 + a];
    }
    __syncthreads();
    if (t < 32) tr[t] += tr[t + 32]; __syncthreads();
    if (t < 16) tr[t] += tr[t + 16]; __syncthreads();
    if (t < 8)  tr[t] += tr[t + 8];  __syncthreads();
    if (t < 4)  tr[t] += tr[t + 4];  __syncthreads();
    if (t == 0) {
        float s2 = 2.0f * (tr[0] + tr[1] + tr[2] + tr[3]);
        norms[k] = fmaxf(sqrtf(fmaxf(s2, 0.0f)), 1e-6f);
    }
}

// ---------- K5: P = B^T A_base (128x128, f32) ----------
__global__ __launch_bounds__(256) void k_P(const float* __restrict__ Af,
                                           const float* __restrict__ Bf,
                                           float* __restrict__ P) {
    int bi = blockIdx.x >> 3, bj = blockIdx.x & 7;
    int i = bi * 16 + (threadIdx.x >> 4);
    int j = bj * 16 + (threadIdx.x & 15);
    float acc = 0.f;
    for (int d = 0; d < D_; ++d)
        acc = fmaf(Bf[(size_t)d * RNK + i], Af[(size_t)d * RNK + j], acc);
    P[(size_t)i * RNK + j] = acc;
}

// ---------- K6a: h = relu(z @ w1 + b1) ----------
__global__ __launch_bounds__(128) void k_h(const float* __restrict__ z,
                                           const float* __restrict__ w1,
                                           const float* __restrict__ b1,
                                           float* __restrict__ h) {
    int b = blockIdx.x, j = threadIdx.x;
    const float* zp = z + (size_t)b * ZD;
    float acc = b1[j];
    for (int i = 0; i < ZD; ++i) acc = fmaf(zp[i], w1[(size_t)i * HID_ + j], acc);
    h[b * HID_ + j] = fmaxf(acc, 0.0f);
}

// ---------- K6b: coeff -> s = coeff/norm ; gate = ADAPTER*sigmoid(.) ----------
__global__ __launch_bounds__(256) void k_cg(const float* __restrict__ h,
                                            const float* __restrict__ wc, const float* __restrict__ bc,
                                            const float* __restrict__ wg, const float* __restrict__ bg,
                                            const float* __restrict__ norms,
                                            float* __restrict__ s, float* __restrict__ gate) {
    int t = threadIdx.x;
    {
        int b = t >> 3, k = t & 7;
        float acc = bc[k];
        for (int j = 0; j < HID_; ++j) acc = fmaf(h[b * HID_ + j], wc[j * K_ + k], acc);
        s[b * K_ + k] = acc * RESIDUAL_SCALE_ / norms[k];
    }
    if (t < 32) {
        float acc = bg[0];
        for (int j = 0; j < HID_; ++j) acc = fmaf(h[t * HID_ + j], wg[j], acc);
        gate[t] = ADAPTER_SCALE_ / (1.0f + expf(-acc));
    }
}

// ---------- K7: per-batch M = I - 0.5*P*colscale; Gauss-Jordan inverse in LDS;
//             write Minv'' = gate * diag(r) * Minv, TRANSPOSED, bf16 ----------
__global__ __launch_bounds__(256) void k_minv(const float* __restrict__ P,
                                              const float* __restrict__ s,
                                              const float* __restrict__ gate,
                                              u16* __restrict__ minvT) {
    extern __shared__ float M[];          // 128 x 129 (padded: conflict-free)
    __shared__ float pinv_s;
    const int LDM = RNK + 1;
    int b = blockIdx.x, t = threadIdx.x;
    const float* sb = s + b * K_;
    for (int e = t; e < RNK * RNK; e += 256) {
        int i = e >> 7, j = e & 127;
        float cs = sb[(j & 63) >> 3];
        M[i * LDM + j] = ((i == j) ? 1.0f : 0.0f) - 0.5f * P[e] * cs;
    }
    __syncthreads();
    int row = t & 127;
    int half = t >> 7;
    int c0 = half * 64;
    for (int p = 0; p < RNK; ++p) {
        float f = M[row * LDM + p];                // old column-p value (own row)
        if (t == 0) pinv_s = 1.0f / M[p * LDM + p];
        __syncthreads();
        float pinv = pinv_s;
        if (row == p)                              // scale pivot row
            for (int c = c0; c < c0 + 64; ++c) M[p * LDM + c] *= pinv;
        __syncthreads();
        if (row != p)                              // eliminate
            for (int c = c0; c < c0 + 64; ++c)
                if (c != p) M[row * LDM + c] = fmaf(-f, M[p * LDM + c], M[row * LDM + c]);
        __syncthreads();
        if (half == 0) {                           // fix column p
            if (row == p) M[p * LDM + p] = pinv;
            else          M[row * LDM + p] = -f * pinv;
        }
        __syncthreads();
    }
    float gb = gate[b];
    for (int e = t; e < RNK * RNK; e += 256) {
        int i = e >> 7, j = e & 127;               // (i,j) of Minv
        float r = gb * sb[(i & 63) >> 3];
        minvT[(size_t)b * RNK * RNK + (size_t)j * RNK + i] = f2bf(r * M[i * LDM + j]);
    }
}

// ---------- WMMA GEMM core: wave computes a 32x128 output tile ----------
// Two A fragments (rows m0, m0+16) reuse every B fragment -> 20 loads per
// 16 WMMAs per K-step instead of 18 per 8.
#define GEMM_TILE_LOOP(ASRC, ALD, BSRC, BLD, KTOT)                               \
    v8f acc[2][8] = {};                                                          \
    for (int kt = 0; kt < (KTOT); kt += 32) {                                    \
        v16bf a0 = load_a16((ASRC), (ALD), m0, kt);                              \
        v16bf a1 = load_a16((ASRC), (ALD), m0 + 16, kt);                         \
        _Pragma("unroll")                                                        \
        for (int tdx = 0; tdx < 8; ++tdx) {                                      \
            v16bf bb = load_bT16((BSRC), (BLD), kt, n0 + tdx * 16);              \
            acc[0][tdx] = __builtin_amdgcn_wmma_f32_16x16x32_bf16(               \
                false, a0, false, bb, (short)0, acc[0][tdx], false, false);      \
            acc[1][tdx] = __builtin_amdgcn_wmma_f32_16x16x32_bf16(               \
                false, a1, false, bb, (short)0, acc[1][tdx], false, false);      \
        }                                                                        \
    }

#define GEMM_STORE_BF16(DST, DLD)                                               \
    {                                                                           \
        int lane = threadIdx.x & 31, hi = lane >> 4, n = lane & 15;             \
        _Pragma("unroll")                                                       \
        for (int rr = 0; rr < 2; ++rr)                                          \
            _Pragma("unroll")                                                   \
            for (int tdx = 0; tdx < 8; ++tdx)                                   \
                _Pragma("unroll")                                               \
                for (int v = 0; v < 8; ++v)                                     \
                    (DST)[(size_t)(m0 + rr * 16 + hi * 8 + v) * (DLD)           \
                          + n0 + tdx * 16 + n] = f2bf(acc[rr][tdx][v]);         \
    }

// Z = x_bf @ A_base   (16384x128, K=768) -> bf16
__global__ __launch_bounds__(256) void k_zgemm(const u16* __restrict__ xbf,
                                               const u16* __restrict__ AbT,
                                               u16* __restrict__ Zbf) {
    int wave = threadIdx.x >> 5;
    int m0 = blockIdx.x * 256 + wave * 32;
    int n0 = 0;
    GEMM_TILE_LOOP(xbf, D_, AbT, D_, D_)
    GEMM_STORE_BF16(Zbf, RNK)
}

// C = W @ B  (768x128, K=768) -> bf16 row-major
__global__ __launch_bounds__(256) void k_cgemm(const u16* __restrict__ Wbf,
                                               const u16* __restrict__ BbT,
                                               u16* __restrict__ Cbf) {
    int wave = threadIdx.x >> 5;
    int m0 = blockIdx.x * 256 + wave * 32;
    int n0 = 0;
    GEMM_TILE_LOOP(Wbf, D_, BbT, D_, D_)
    GEMM_STORE_BF16(Cbf, RNK)
}

// Y = Z @ Minv''_b  (per-batch B operand; 16384x128, K=128) -> bf16
__global__ __launch_bounds__(256) void k_ygemm(const u16* __restrict__ Zbf,
                                               const u16* __restrict__ minvT,
                                               u16* __restrict__ Ybf) {
    int wave = threadIdx.x >> 5;
    int m0 = blockIdx.x * 256 + wave * 32;
    int n0 = 0;
    const u16* BT = minvT + (size_t)(blockIdx.x >> 1) * RNK * RNK;  // 2 blocks per batch (256 rows)
    GEMM_TILE_LOOP(Zbf, RNK, BT, RNK, RNK)
    GEMM_STORE_BF16(Ybf, RNK)
}

// out = x@W^T + bias + Y@C^T   (fused epilogue; gate already folded into Y)
__global__ __launch_bounds__(256) void k_out(const u16* __restrict__ xbf,
                                             const u16* __restrict__ Wbf,
                                             const u16* __restrict__ Ybf,
                                             const u16* __restrict__ Cbf,
                                             const float* __restrict__ bias,
                                             float* __restrict__ out) {
    int wave = threadIdx.x >> 5;
    int m0 = blockIdx.x * 256 + wave * 32;
    int n0 = blockIdx.y * 128;
    v8f acc[2][8] = {};
    for (int kt = 0; kt < D_; kt += 32) {            // base GEMM, K=768
        v16bf a0 = load_a16(xbf, D_, m0, kt);
        v16bf a1 = load_a16(xbf, D_, m0 + 16, kt);
#pragma unroll
        for (int tdx = 0; tdx < 8; ++tdx) {
            v16bf bb = load_bT16(Wbf, D_, kt, n0 + tdx * 16);   // W row-major == (W^T) transposed source
            acc[0][tdx] = __builtin_amdgcn_wmma_f32_16x16x32_bf16(
                false, a0, false, bb, (short)0, acc[0][tdx], false, false);
            acc[1][tdx] = __builtin_amdgcn_wmma_f32_16x16x32_bf16(
                false, a1, false, bb, (short)0, acc[1][tdx], false, false);
        }
    }
    for (int kt = 0; kt < RNK; kt += 32) {           // low-rank delta, K=128
        v16bf a0 = load_a16(Ybf, RNK, m0, kt);
        v16bf a1 = load_a16(Ybf, RNK, m0 + 16, kt);
#pragma unroll
        for (int tdx = 0; tdx < 8; ++tdx) {
            v16bf bb = load_bT16(Cbf, RNK, kt, n0 + tdx * 16);  // C row-major == (C^T) transposed source
            acc[0][tdx] = __builtin_amdgcn_wmma_f32_16x16x32_bf16(
                false, a0, false, bb, (short)0, acc[0][tdx], false, false);
            acc[1][tdx] = __builtin_amdgcn_wmma_f32_16x16x32_bf16(
                false, a1, false, bb, (short)0, acc[1][tdx], false, false);
        }
    }
    int lane = threadIdx.x & 31, hi = lane >> 4, n = lane & 15;
#pragma unroll
    for (int rr = 0; rr < 2; ++rr)
#pragma unroll
        for (int tdx = 0; tdx < 8; ++tdx) {
            float bv = bias[n0 + tdx * 16 + n];
#pragma unroll
            for (int v = 0; v < 8; ++v)
                out[(size_t)(m0 + rr * 16 + hi * 8 + v) * D_ + n0 + tdx * 16 + n]
                    = acc[rr][tdx][v] + bv;
        }
}

// ---------- launch ----------
extern "C" void kernel_launch(void* const* d_in, const int* in_sizes, int n_in,
                              void* d_out, int out_size, void* d_ws, size_t ws_size,
                              hipStream_t stream) {
    (void)in_sizes; (void)n_in; (void)out_size; (void)ws_size;
    const float* x    = (const float*)d_in[0];
    const float* W    = (const float*)d_in[1];
    const float* bias = (const float*)d_in[2];
    const float* lb   = (const float*)d_in[3];
    const float* rb   = (const float*)d_in[4];
    const float* w1   = (const float*)d_in[5];
    const float* b1   = (const float*)d_in[6];
    const float* wc   = (const float*)d_in[7];
    const float* bc   = (const float*)d_in[8];
    const float* wg   = (const float*)d_in[9];
    const float* bg   = (const float*)d_in[10];

    char* ws = (char*)d_ws;
    u16*   xbf   = (u16*)(ws + OFF_XBF);
    float* zbuf  = (float*)(ws + OFF_Z);
    float* hbuf  = (float*)(ws + OFF_H);
    float* norms = (float*)(ws + OFF_NRM);
    float* gateb = (float*)(ws + OFF_GATE);
    float* sbuf  = (float*)(ws + OFF_S);
    float* Af    = (float*)(ws + OFF_AF);
    float* Bf    = (float*)(ws + OFF_BF);
    u16*   AbT   = (u16*)(ws + OFF_ABT);
    u16*   BbT   = (u16*)(ws + OFF_BBT);
    u16*   Wbf   = (u16*)(ws + OFF_WBF);
    float* Pbuf  = (float*)(ws + OFF_P);
    u16*   minvT = (u16*)(ws + OFF_MIT);
    u16*   Zbf   = (u16*)(ws + OFF_ZBF);
    u16*   Ybf   = (u16*)(ws + OFF_YBF);
    u16*   Cbf   = (u16*)(ws + OFF_CBF);

    k_stats<<<(B_ * D_) / 256, 256, 0, stream>>>(x, zbuf, xbf);
    k_build<<<(D_ * RNK) / 256, 256, 0, stream>>>(lb, rb, Af, Bf, AbT, BbT);
    k_wbf  <<<(D_ * D_) / 256, 256, 0, stream>>>(W, Wbf);
    k_norm <<<K_, 256, 0, stream>>>(lb, rb, norms);
    k_P    <<<64, 256, 0, stream>>>(Af, Bf, Pbuf);
    k_h    <<<B_, 128, 0, stream>>>(zbuf, w1, b1, hbuf);
    k_cg   <<<1, 256, 0, stream>>>(hbuf, wc, bc, wg, bg, norms, sbuf, gateb);
    k_minv <<<B_, 256, (size_t)RNK * (RNK + 1) * 4, stream>>>(Pbuf, sbuf, gateb, minvT);
    k_zgemm<<<MTOT / 256, 256, 0, stream>>>(xbf, AbT, Zbf);
    k_cgemm<<<D_ / 256, 256, 0, stream>>>(Wbf, BbT, Cbf);
    k_ygemm<<<MTOT / 256, 256, 0, stream>>>(Zbf, minvT, Ybf);
    k_out  <<<dim3(MTOT / 256, D_ / 128), 256, 0, stream>>>(xbf, Wbf, Ybf, Cbf, bias, (float*)d_out);
}
